// CF_spikes_41884521070633
// MI455X (gfx1250) — matrix operand order repaired and checked by reading, not compiled, not run on hardware
//
#include <hip/hip_runtime.h>

typedef __attribute__((ext_vector_type(16))) _Float16 v16h;
typedef __attribute__((ext_vector_type(8)))  _Float16 v8h;
typedef __attribute__((ext_vector_type(8)))  float    v8f;

namespace {

constexpr int kDuration = 16;
constexpr int kBTile    = 64;    // batch rows per block
constexpr int kIn       = 784;
constexpr int kKcA      = 13;    // first K-half: chunks [0,13)  -> cols [0,416)
constexpr int kKcB      = 12;    // second K-half: chunks [13,25) -> cols [416,800)
constexpr int kColsA    = kKcA * 32;   // 416
constexpr int kColsB    = kKcB * 32;   // 384
constexpr int kH1       = 128;
constexpr int kH2       = 64;
constexpr int kOut      = 10;
constexpr int kOutPad   = 16;

// LDS row strides in halves; multiples of 8 keep 16B alignment for b128 and
// perturb bank mapping away from pure power-of-two strides.
constexpr int kW1Stride = 808;   // W1 LDS: [128][808]
constexpr int kS0Stride = 424;   // half-K spike buffer: [64][424]
constexpr int kS1Stride = 136;
constexpr int kS2Stride = 72;
constexpr int kW2Stride = 136;   // [64][136]
constexpr int kW3Stride = 72;    // [16][72]

__device__ __forceinline__ v16h concat8(v8h lo, v8h hi) {
  return __builtin_shufflevector(lo, hi, 0, 1, 2, 3, 4, 5, 6, 7,
                                         8, 9, 10, 11, 12, 13, 14, 15);
}

// A fragment (16x32 f16): lanes 0-15 hold M=lane, halves [kc*32+0..7] and
// [kc*32+16..23]; lanes 16-31 hold M=lane-16, halves [+8..15] and [+24..31].
__device__ __forceinline__ v16h load_a(const _Float16* base, int stride,
                                       int mrow, int kc, int lane) {
  const int m  = mrow + (lane & 15);
  const int kh = (lane >> 4) << 3;              // 0 or 8 halves
  const _Float16* p = base + m * stride + kc * 32 + kh;
  return concat8(*(const v8h*)p, *(const v8h*)(p + 16));
}

// B fragment (32x16 f16), B(k,n) = W[n][k] with W row-major [N][K-stride]:
// lanes 0-15: N=lane, K=kc*32+0..7,16..23; lanes 16-31: N=lane-16, K=kc*32+8..15,24..31.
__device__ __forceinline__ v16h load_b(const _Float16* base, int stride,
                                       int ncol, int kc, int lane) {
  const int n = ncol + (lane & 15);
  const int k = kc * 32 + ((lane >> 4) << 3);   // 0 or 8 halves
  const _Float16* p = base + n * stride + k;
  return concat8(*(const v8h*)p, *(const v8h*)(p + 16));
}

__device__ __forceinline__ v8f wmma_f16(v16h a, v16h b, v8f c) {
  return __builtin_amdgcn_wmma_f32_16x16x32_f16(
      /*neg_a=*/false, a, /*neg_b=*/false, b,
      /*c_mod=*/(short)0, c, /*reuse_a=*/false, /*reuse_b=*/false);
}

// Add bias, fire, soft-reset; write f16 spikes to LDS using the 16x16 f32
// C/D layout: VGPR r -> M = r + 8*(lane>>4), N = lane&15.
__device__ __forceinline__ void fire_store(v8f& v, float bias, _Float16* dst,
                                           int stride, int mrow, int ncol,
                                           int lane) {
  const int n  = ncol + (lane & 15);
  const int mb = mrow + ((lane >> 4) << 3);
#pragma unroll
  for (int r = 0; r < 8; ++r) {
    float x = v[r] + bias;
    float s = (x >= 1.0f) ? 1.0f : 0.0f;
    v[r] = x - s;
    dst[(mb + r) * stride + n] = (_Float16)s;
  }
}

}  // namespace

// 256 threads = 8 wave32s on one WGP (4 SIMD32s) -> the block needs 2 waves
// per SIMD to launch; tell the compiler so it caps VGPRs accordingly.
__global__ __launch_bounds__(256, 2) void snn_wmma_kernel(
    const float* __restrict__ features, const float* __restrict__ w1,
    const float* __restrict__ w2, const float* __restrict__ w3,
    const float* __restrict__ b1, const float* __restrict__ b2,
    const float* __restrict__ b3, const float* __restrict__ out_scale,
    float* __restrict__ out) {
  // 300.25 KB total — LDS-resident weights (read from global exactly once
  // per block); steady-state global traffic is the feature tile only.
  __shared__ _Float16 w1s[kH1 * kW1Stride];      // 206,848 B
  __shared__ _Float16 s0a[kBTile * kS0Stride];   //  54,272 B (half-K spikes)
  __shared__ _Float16 s1s[kBTile * kS1Stride];   //  17,408 B
  __shared__ _Float16 s2s[kBTile * kS2Stride];   //   9,216 B
  __shared__ _Float16 w2s[kH2 * kW2Stride];      //  17,408 B
  __shared__ _Float16 w3s[kOutPad * kW3Stride];  //   2,304 B

  const int tid  = threadIdx.x;
  const int lane = tid & 31;
  const int w    = tid >> 5;  // wave id 0..7 (wave32)
  const int row0 = blockIdx.x * kBTile;

  // ---- One-time: stage weights in LDS (fp32 -> f16, zero-padded) ----
  for (int i = tid; i < kH1 * kW1Stride; i += 256) {
    const int n = i / kW1Stride, k = i - n * kW1Stride;
    w1s[i] = (_Float16)((k < kIn) ? w1[n * kIn + k] : 0.0f);
  }
  for (int i = tid; i < kH2 * kW2Stride; i += 256) {
    const int n = i / kW2Stride, k = i - n * kW2Stride;
    w2s[i] = (_Float16)((k < kH1) ? w2[n * kH1 + k] : 0.0f);
  }
  for (int i = tid; i < kOutPad * kW3Stride; i += 256) {
    const int n = i / kW3Stride, k = i - n * kW3Stride;
    w3s[i] = (_Float16)((n < kOut && k < kH2) ? w3[n * kH2 + k] : 0.0f);
  }
  // s1s/s2s padding columns must be zero for WMMA reads; real columns are
  // rewritten every step, pads stay zero forever.
  for (int i = tid; i < kBTile * kS1Stride; i += 256) s1s[i] = (_Float16)0.0f;
  for (int i = tid; i < kBTile * kS2Stride; i += 256) s2s[i] = (_Float16)0.0f;

  // Layer1: wave -> M-tiles {mt0, mt0+16} x N-tiles {nt1, nt1+16} (64x128 out).
  const int mt0 = (w & 1) * 32;
  const int nt1 = (w >> 1) * 32;
  // Layer2: wave -> N-tile nt2, M-tiles {mt2, mt2+16} (64x64 out).
  const int nt2 = (w & 3) * 16;
  const int mt2 = (w >> 2) * 32;
  // Layer3: waves 0..3 -> M-tile w*16 (64x16 out).
  const int mt3 = w * 16;

  const int   nl   = lane & 15;
  const float b1v0 = b1[nt1 + nl];
  const float b1v1 = b1[nt1 + 16 + nl];
  const float b2v  = b2[nt2 + nl];
  float b3v = 0.0f, scv = 0.0f;
  if (nl < kOut) { b3v = b3[nl]; scv = out_scale[nl]; }

  // Membranes live in WMMA accumulator fragments for all 16 steps.
  v8f v1f[4] = {};
  v8f v2f[2] = {};
  v8f v3f    = {};
  v8f cnt    = {};

  __syncthreads();

  for (int t = 1; t <= kDuration; ++t) {
    const float tf = (float)t;

    // ---- Encoder half A: cols [0,416). Stateless rate coding: for
    // f in [0,1), THRESH=1 the spike at step t is floor(t*f)-floor((t-1)*f).
    for (int idx = tid; idx < kBTile * kColsA; idx += 256) {
      const int r = idx / kColsA;
      const int c = idx - r * kColsA;
      const float f = features[(size_t)(row0 + r) * kIn + c];
      const float s = floorf(tf * f) - floorf((tf - 1.0f) * f);
      s0a[r * kS0Stride + c] = (_Float16)s;
    }
    __syncthreads();

    // ---- Layer 1, K-chunks [0,13)
    for (int kc = 0; kc < kKcA; ++kc) {
      const v16h bf0 = load_b(w1s, kW1Stride, nt1, kc, lane);
      const v16h bf1 = load_b(w1s, kW1Stride, nt1 + 16, kc, lane);
      const v16h af0 = load_a(s0a, kS0Stride, mt0, kc, lane);
      const v16h af1 = load_a(s0a, kS0Stride, mt0 + 16, kc, lane);
      v1f[0] = wmma_f16(af0, bf0, v1f[0]);
      v1f[1] = wmma_f16(af0, bf1, v1f[1]);
      v1f[2] = wmma_f16(af1, bf0, v1f[2]);
      v1f[3] = wmma_f16(af1, bf1, v1f[3]);
    }
    __syncthreads();

    // ---- Encoder half B: cols [416,800); cols >= 784 are zero padding.
    for (int idx = tid; idx < kBTile * kColsB; idx += 256) {
      const int r = idx / kColsB;
      const int c = idx - r * kColsB;        // local col, global = c + 416
      const int cg = c + kColsA;
      float s = 0.0f;
      if (cg < kIn) {
        const float f = features[(size_t)(row0 + r) * kIn + cg];
        s = floorf(tf * f) - floorf((tf - 1.0f) * f);
      }
      s0a[r * kS0Stride + c] = (_Float16)s;
    }
    __syncthreads();

    // ---- Layer 1, K-chunks [13,25): spikes at local chunk kc, weights at
    // global chunk kc+13.
    for (int kc = 0; kc < kKcB; ++kc) {
      const v16h bf0 = load_b(w1s, kW1Stride, nt1, kc + kKcA, lane);
      const v16h bf1 = load_b(w1s, kW1Stride, nt1 + 16, kc + kKcA, lane);
      const v16h af0 = load_a(s0a, kS0Stride, mt0, kc, lane);
      const v16h af1 = load_a(s0a, kS0Stride, mt0 + 16, kc, lane);
      v1f[0] = wmma_f16(af0, bf0, v1f[0]);
      v1f[1] = wmma_f16(af0, bf1, v1f[1]);
      v1f[2] = wmma_f16(af1, bf0, v1f[2]);
      v1f[3] = wmma_f16(af1, bf1, v1f[3]);
    }
    fire_store(v1f[0], b1v0, s1s, kS1Stride, mt0,      nt1,      lane);
    fire_store(v1f[1], b1v1, s1s, kS1Stride, mt0,      nt1 + 16, lane);
    fire_store(v1f[2], b1v0, s1s, kS1Stride, mt0 + 16, nt1,      lane);
    fire_store(v1f[3], b1v1, s1s, kS1Stride, mt0 + 16, nt1 + 16, lane);
    __syncthreads();

    // ---- Layer 2
    for (int kc = 0; kc < kH1 / 32; ++kc) {
      const v16h bf  = load_b(w2s, kW2Stride, nt2, kc, lane);
      const v16h af0 = load_a(s1s, kS1Stride, mt2, kc, lane);
      const v16h af1 = load_a(s1s, kS1Stride, mt2 + 16, kc, lane);
      v2f[0] = wmma_f16(af0, bf, v2f[0]);
      v2f[1] = wmma_f16(af1, bf, v2f[1]);
    }
    fire_store(v2f[0], b2v, s2s, kS2Stride, mt2,      nt2, lane);
    fire_store(v2f[1], b2v, s2s, kS2Stride, mt2 + 16, nt2, lane);
    __syncthreads();

    // ---- Layer 3 (waves 0..3; wave-uniform branch keeps EXEC all-ones)
    if (w < 4) {
      for (int kc = 0; kc < kH2 / 32; ++kc) {
        const v16h bf = load_b(w3s, kW3Stride, 0, kc, lane);
        const v16h af = load_a(s2s, kS2Stride, mt3, kc, lane);
        v3f = wmma_f16(af, bf, v3f);
      }
#pragma unroll
      for (int r = 0; r < 8; ++r) {
        float x = v3f[r] + b3v;
        float s = (x >= 1.0f) ? 1.0f : 0.0f;
        v3f[r] = x - s;
        cnt[r] += s;
      }
    }
    __syncthreads();
  }

  // reconstruct: out = (counts / DURATION) * out_scale, only real 10 columns.
  if (w < 4 && nl < kOut) {
    const int   mb = row0 + mt3 + ((lane >> 4) << 3);
    const float sc = scv * (1.0f / (float)kDuration);
#pragma unroll
    for (int r = 0; r < 8; ++r) {
      out[(size_t)(mb + r) * kOut + nl] = cnt[r] * sc;
    }
  }
}

extern "C" void kernel_launch(void* const* d_in, const int* in_sizes, int n_in,
                              void* d_out, int out_size, void* d_ws,
                              size_t ws_size, hipStream_t stream) {
  const float* features  = (const float*)d_in[0];
  const float* w1        = (const float*)d_in[1];
  const float* b1        = (const float*)d_in[2];
  const float* w2        = (const float*)d_in[3];
  const float* b2        = (const float*)d_in[4];
  const float* w3        = (const float*)d_in[5];
  const float* b3        = (const float*)d_in[6];
  const float* out_scale = (const float*)d_in[7];

  const int batch = in_sizes[0] / kIn;  // 32768
  snn_wmma_kernel<<<batch / kBTile, 256, 0, stream>>>(
      features, w1, w2, w3, b1, b2, b3, out_scale, (float*)d_out);
}
